// encoder_21758304322093
// MI455X (gfx1250) — matrix operand order repaired and checked by reading, not compiled, not run on hardware
//
#include <hip/hip_runtime.h>

#define B_    1024
#define TM1   255
#define D_    128
#define H_    128
#define NG    512      // 4*H gate columns
#define WSTRIDE 264    // padded halves per weight row (16B-aligned, bank spread)
#define ASTRIDE 264    // padded halves per A-tile row

typedef __bf16 bf16_t;
typedef bf16_t v16bf __attribute__((ext_vector_type(16)));
typedef float  v8f   __attribute__((ext_vector_type(8)));

union Frag {
    v16bf v;
    uint4 q[2];
};

__device__ __forceinline__ unsigned short f2bf(float f) {
    unsigned int u = __float_as_uint(f);
    u += 0x7FFFu + ((u >> 16) & 1u);          // round-to-nearest-even
    return (unsigned short)(u >> 16);
}
__device__ __forceinline__ unsigned int pack2bf(float a, float b) {
    return (unsigned int)f2bf(a) | ((unsigned int)f2bf(b) << 16);
}
__device__ __forceinline__ float sigmoidf_(float x) {
    float xc = fminf(fmaxf(x, -30.f), 30.f);
    return __fdividef(1.f, 1.f + __expf(-xc));
}
__device__ __forceinline__ float tanh_fast(float x) {
    float xc = fminf(fmaxf(x, -15.f), 15.f);
    float e  = __expf(2.f * xc);
    return __fdividef(e - 1.f, e + 1.f);
}

// LDS-only barrier: publish DS traffic, do NOT wait for global store acks
// (out_wx / out_h stores drain in the background; s_endpgm waits at the end).
#define LDS_BARRIER()                                                        \
    asm volatile("s_wait_dscnt 0x0\n\t"                                      \
                 "s_barrier_signal -1\n\t"                                   \
                 "s_barrier_wait -1" ::: "memory")

// ---------------------------------------------------------------------------
// Kernel 1: attn[b,d] = softmax_d( sum_t x[b,t,d] * Wx[t] )
// (h.Wh + c.Wc + ba add a per-row constant -> cancels in softmax)
// ---------------------------------------------------------------------------
__global__ void attn_precompute(const float* __restrict__ x,
                                const float* __restrict__ Wa,
                                float* __restrict__ attn_out) {
    __shared__ float wx_s[TM1];
    __shared__ float red[4];
    const int b   = blockIdx.x;
    const int tid = threadIdx.x;           // 0..127

    for (int i = tid; i < TM1; i += 128) wx_s[i] = Wa[2 * H_ + i];
    __syncthreads();

    const float* xb = x + (size_t)b * TM1 * D_ + tid;
    float s = 0.f;
    for (int t = 0; t < TM1; ++t) s += xb[(size_t)t * D_] * wx_s[t];

    float m = s;
    for (int off = 16; off >= 1; off >>= 1) m = fmaxf(m, __shfl_xor(m, off, 32));
    if ((tid & 31) == 0) red[tid >> 5] = m;
    __syncthreads();
    m = fmaxf(fmaxf(red[0], red[1]), fmaxf(red[2], red[3]));

    float e = __expf(s - m);
    float ss = e;
    for (int off = 16; off >= 1; off >>= 1) ss += __shfl_xor(ss, off, 32);
    __syncthreads();
    if ((tid & 31) == 0) red[tid >> 5] = ss;
    __syncthreads();
    ss = red[0] + red[1] + red[2] + red[3];

    attn_out[(size_t)b * D_ + tid] = e / ss;
}

// ---------------------------------------------------------------------------
// Kernel 2: persistent recurrence. 64 blocks x 16 batch rows, 256 thr = 8 waves.
// LDS: W=[W_ih|W_hh] bf16 (512x264 halves), A=[wx|h] (16x264 halves),
//      x double-buffer stage (2 x 16x128 f32) fed by global_load_async_to_lds.
// c-state, attn, gate accumulators live in VGPRs.
// Wave w owns gate columns { w*16+j, +128, +256, +384 } -> i,f,g,o in regs.
// ---------------------------------------------------------------------------
__global__ void __launch_bounds__(256)
encoder_main(const float* __restrict__ x,
             const float* __restrict__ W_ih,
             const float* __restrict__ W_hh,
             const float* __restrict__ b_ih,
             const float* __restrict__ b_hh,
             const float* __restrict__ attn_g,
             float* __restrict__ out_wx,
             float* __restrict__ out_h) {
    extern __shared__ unsigned char smem_raw[];
    unsigned short* Wbf = (unsigned short*)smem_raw;          // 512*264 halves
    unsigned short* A_s = Wbf + NG * WSTRIDE;                 // 16*264 halves
    float* xst          = (float*)(A_s + 16 * ASTRIDE);       // 2*16*128 f32

    const int tid  = threadIdx.x;
    const int lane = tid & 31;
    const int wave = tid >> 5;     // 0..7
    const int ln   = lane & 15;
    const int hi   = lane >> 4;
    const int bm   = blockIdx.x * 16;

    // byte offset of x stage inside the (dynamic-only) LDS segment
    const unsigned xstage_base =
        (unsigned)((NG * WSTRIDE + 16 * ASTRIDE) * sizeof(unsigned short));

    // ---- stage combined weights into LDS as bf16 (one-time) ----
    for (int k = 0; k < 128; ++k) {
        int idx4 = tid + k * 256;          // 32768 float4 total
        int n    = idx4 >> 6;              // weight row 0..511
        int kk   = (idx4 & 63) << 2;       // 0..252
        float4 v = (kk < 128)
            ? *(const float4*)(W_ih + (size_t)n * 128 + kk)
            : *(const float4*)(W_hh + (size_t)n * 128 + (kk - 128));
        unsigned int* dst = (unsigned int*)(Wbf + n * WSTRIDE + kk);
        dst[0] = pack2bf(v.x, v.y);
        dst[1] = pack2bf(v.z, v.w);
    }

    // this thread's two (m,d) slots of the 16x128 x/wx tile
    const int mA = tid >> 5;
    const int dA = (tid & 31) << 2;
    const int mB = mA + 8;
    const int dB = dA;

    // ---- attn values this thread needs, straight to registers ----
    const float4 avA = *(const float4*)(attn_g + (size_t)(bm + mA) * D_ + dA);
    const float4 avB = *(const float4*)(attn_g + (size_t)(bm + mB) * D_ + dB);

    // ---- zero A h-part ----
    {
        unsigned int* hz = (unsigned int*)(A_s + mA * ASTRIDE + 128 + dA);
        hz[0] = 0u; hz[1] = 0u;
        hz = (unsigned int*)(A_s + mB * ASTRIDE + 128 + dB);
        hz[0] = 0u; hz[1] = 0u;
    }

    // ---- per-lane gate columns + fused bias; c-state in regs ----
    float bias[4];
    int   ncol[4];
#pragma unroll
    for (int tt = 0; tt < 4; ++tt) {
        int n   = wave * 16 + ln + tt * 128;
        ncol[tt] = n;
        bias[tt] = b_ih[n] + b_hh[n];
    }
    float creg[8];
#pragma unroll
    for (int r = 0; r < 8; ++r) creg[r] = 0.f;

    const int jg = wave * 16 + ln;         // owned h column

    // ---- preamble: async-stage x_0 into buffer 0 ----
    {
        unsigned lA = xstage_base + (unsigned)((mA * 128 + dA) << 2);
        unsigned lB = xstage_base + (unsigned)((mB * 128 + dB) << 2);
        unsigned long long gA =
            (unsigned long long)(uintptr_t)(x + ((size_t)(bm + mA) * TM1 + 0) * D_ + dA);
        unsigned long long gB =
            (unsigned long long)(uintptr_t)(x + ((size_t)(bm + mB) * TM1 + 0) * D_ + dB);
        asm volatile("global_load_async_to_lds_b128 %0, %1, off"
                     :: "v"(lA), "v"(gA) : "memory");
        asm volatile("global_load_async_to_lds_b128 %0, %1, off"
                     :: "v"(lB), "v"(gB) : "memory");
    }

    __syncthreads();                       // one-time: W + A h-part staged

    for (int t = 0; t < TM1; ++t) {
        const int buf = t & 1;

        // ---- x_t has landed in LDS (this wave's own async loads) ----
        asm volatile("s_wait_asynccnt 0x0" ::: "memory");

        // ---- wx = attn * x_t ; store wx out and into A[:,0:128] (bf16) ----
        {
            const float* xs = xst + buf * 2048;
            float4 xvA = *(const float4*)(xs + mA * 128 + dA);
            float4 xvB = *(const float4*)(xs + mB * 128 + dB);
            float4 wA, wB;
            wA.x = avA.x * xvA.x; wA.y = avA.y * xvA.y;
            wA.z = avA.z * xvA.z; wA.w = avA.w * xvA.w;
            wB.x = avB.x * xvB.x; wB.y = avB.y * xvB.y;
            wB.z = avB.z * xvB.z; wB.w = avB.w * xvB.w;
            size_t gA = ((size_t)(bm + mA) * TM1 + t) * D_ + dA;
            size_t gB = ((size_t)(bm + mB) * TM1 + t) * D_ + dB;
            *(float4*)(out_wx + gA) = wA;
            *(float4*)(out_wx + gB) = wB;
            unsigned int* dst = (unsigned int*)(A_s + mA * ASTRIDE + dA);
            dst[0] = pack2bf(wA.x, wA.y);
            dst[1] = pack2bf(wA.z, wA.w);
            dst = (unsigned int*)(A_s + mB * ASTRIDE + dB);
            dst[0] = pack2bf(wB.x, wB.y);
            dst[1] = pack2bf(wB.z, wB.w);
        }

        // ---- issue async stage of x_{t+1} into the other buffer ----
        {
            const int tn = (t + 1 < TM1) ? t + 1 : t;
            unsigned lA = xstage_base + (unsigned)((((1 - buf) * 2048) + mA * 128 + dA) << 2);
            unsigned lB = xstage_base + (unsigned)((((1 - buf) * 2048) + mB * 128 + dB) << 2);
            unsigned long long gA =
                (unsigned long long)(uintptr_t)(x + ((size_t)(bm + mA) * TM1 + tn) * D_ + dA);
            unsigned long long gB =
                (unsigned long long)(uintptr_t)(x + ((size_t)(bm + mB) * TM1 + tn) * D_ + dB);
            asm volatile("global_load_async_to_lds_b128 %0, %1, off"
                         :: "v"(lA), "v"(gA) : "memory");
            asm volatile("global_load_async_to_lds_b128 %0, %1, off"
                         :: "v"(lB), "v"(gB) : "memory");
        }

        LDS_BARRIER();     // A tile visible to all waves; global stores not held

        // ---- gates = [wx|h] @ Wt : k-chunk 0 starts from C = 0 (inline) ----
        v8f acc[4];
#pragma unroll
        for (int kc = 0; kc < 8; ++kc) {
            Frag a;   // A 16x32 bf16: lanes0-15 K=[0..7|16..23], lanes16-31 +8
            const unsigned short* ap = A_s + ln * ASTRIDE + kc * 32 + hi * 8;
            a.q[0] = *(const uint4*)(ap);
            a.q[1] = *(const uint4*)(ap + 16);
            Frag bfr[4];
#pragma unroll
            for (int tt = 0; tt < 4; ++tt) {
                const unsigned short* bp =
                    Wbf + ncol[tt] * WSTRIDE + kc * 32 + hi * 8;
                bfr[tt].q[0] = *(const uint4*)(bp);
                bfr[tt].q[1] = *(const uint4*)(bp + 16);
            }
            if (kc == 0) {
#pragma unroll
                for (int tt = 0; tt < 4; ++tt)
                    acc[tt] = __builtin_amdgcn_wmma_f32_16x16x32_bf16(
                        false, a.v, false, bfr[tt].v,
                        (short)0, (v8f)(0.f), false, false);
            } else {
#pragma unroll
                for (int tt = 0; tt < 4; ++tt)
                    acc[tt] = __builtin_amdgcn_wmma_f32_16x16x32_bf16(
                        false, a.v, false, bfr[tt].v,
                        (short)0, acc[tt], false, false);
            }
        }

        LDS_BARRIER();     // all waves done reading A before h-part rewrite

        // ---- LSTM pointwise in registers; bias folded in here ----
#pragma unroll
        for (int r = 0; r < 8; ++r) {
            int m = r + 8 * hi;            // C/D layout: M = vgpr + 8*(lane>=16)
            float iv = sigmoidf_(acc[0][r] + bias[0]);
            float fv = sigmoidf_(acc[1][r] + bias[1]);
            float gv = tanh_fast(acc[2][r] + bias[2]);
            float ov = sigmoidf_(acc[3][r] + bias[3]);
            float cn = fv * creg[r] + iv * gv;
            float hn = ov * tanh_fast(cn);
            creg[r] = cn;
            out_h[((size_t)(bm + m) * TM1 + t) * H_ + jg] = hn;
            A_s[m * ASTRIDE + 128 + jg] = f2bf(hn);   // h for next step's GEMM
        }
        // h-part writes are disjoint from next wx writes; the next pre-GEMM
        // barrier publishes them.
    }
}

extern "C" void kernel_launch(void* const* d_in, const int* in_sizes, int n_in,
                              void* d_out, int out_size, void* d_ws, size_t ws_size,
                              hipStream_t stream) {
    const float* input_data = (const float*)d_in[0];
    const float* Wa         = (const float*)d_in[1];
    // d_in[2] = ba : cancels inside softmax, unused
    const float* W_ih       = (const float*)d_in[3];
    const float* W_hh       = (const float*)d_in[4];
    const float* b_ih       = (const float*)d_in[5];
    const float* b_hh       = (const float*)d_in[6];

    float* out_wx = (float*)d_out;                                 // (B,TM1,D)
    float* out_h  = out_wx + (size_t)B_ * TM1 * D_;                // (B,TM1,H)
    float* attn   = (float*)d_ws;                                  // B*D f32

    attn_precompute<<<B_, 128, 0, stream>>>(input_data, Wa, attn);

    size_t lds = (size_t)(NG * WSTRIDE + 16 * ASTRIDE) * sizeof(unsigned short)
               + (size_t)2 * 16 * 128 * sizeof(float);             // ~295 KB
    encoder_main<<<B_ / 16, 256, lds, stream>>>(input_data, W_ih, W_hh,
                                                b_ih, b_hh, attn,
                                                out_wx, out_h);
}
